// DecoderAttention_67199058313878
// MI455X (gfx1250) — matrix operand order repaired and checked by reading, not compiled, not run on hardware
//
#include <hip/hip_runtime.h>

typedef __attribute__((ext_vector_type(16))) _Float16 v16h;
typedef __attribute__((ext_vector_type(8)))  float    v8f;

#define B_   128
#define T_   2048
#define H_   200
#define H4_  800
#define NT_  13   // N tiles of 16 covering H=200 (padded to 208)
#define NK_  7    // K steps of 32 covering H=200 (padded to 224)
#define BPACK_BYTES (NT_ * NK_ * 32 * 16 * 2)   // 93184
#define BPACK_VEC4  (BPACK_BYTES / 16)          // 5824

// workspace layout (bytes)
#define WS_BPACK   0                      // 93184 B
#define WS_QB      93184                  // 128*200 f32
#define WS_HPRE    (WS_QB + 102400)       // 128*800 f32
#define WS_SCORES  (WS_HPRE + 409600)     // 128*2048 f32
#define WS_CTX     (WS_SCORES + 1048576)  // 128*200 f32

// Branch-free tanh: lowers to v_exp_f32 + v_rcp_f32, no EXEC divergence.
__device__ __forceinline__ float fast_tanh(float x) {
  return 1.f - 2.f / (__expf(2.f * x) + 1.f);
}
__device__ __forceinline__ float fast_sigmoid(float x) {
  return 1.f / (1.f + __expf(-x));
}

// ---------------------------------------------------------------------------
// 1) qproj[b,h] = h0[b]@Wa.T + ba + bua   ;  hpre[b,j] = h0[b]@W_hh.T + b_hh
// ---------------------------------------------------------------------------
__global__ __launch_bounds__(256) void k_prep(
    const float* __restrict__ h0, const float* __restrict__ Wa,
    const float* __restrict__ ba, const float* __restrict__ bua,
    const float* __restrict__ Whh, const float* __restrict__ bhh,
    float* __restrict__ qb, float* __restrict__ hpre) {
  __shared__ float q[H_];
  int b = blockIdx.x, tid = threadIdx.x;
  for (int k = tid; k < H_; k += 256) q[k] = h0[b * H_ + k];
  __syncthreads();
  for (int h = tid; h < H_; h += 256) {
    float acc = ba[h] + bua[h];
    const float* w = Wa + h * H_;
    for (int k = 0; k < H_; ++k) acc += q[k] * w[k];
    qb[b * H_ + h] = acc;
  }
  for (int j = tid; j < H4_; j += 256) {
    float acc = bhh[j];
    const float* w = Whh + j * H_;
    for (int k = 0; k < H_; ++k) acc += q[k] * w[k];
    hpre[b * H4_ + j] = acc;
  }
}

// ---------------------------------------------------------------------------
// 2) Pack Ua into WMMA B-fragment layout (f16), zero-padded to 224x208.
//    B[k][n] = Ua[n][k].  Lane<16: elem e -> K=32kk+e, N=lane.
//    Lane>=16: elem e -> K=32kk+16+e, N=lane-16.
// ---------------------------------------------------------------------------
__global__ __launch_bounds__(256) void k_pack(const float* __restrict__ Ua,
                                              _Float16* __restrict__ bp) {
  int idx = blockIdx.x * blockDim.x + threadIdx.x;
  const int total = NT_ * NK_ * 32 * 16;
  if (idx >= total) return;
  int e    = idx & 15;
  int lane = (idx >> 4) & 31;
  int g    = idx >> 9;          // nt*7 + kk
  int kk   = g % NK_;
  int nt   = g / NK_;
  int n = nt * 16 + (lane & 15);
  int k = 32 * kk + 16 * (lane >> 4) + e;
  float v = (n < H_ && k < H_) ? Ua[n * H_ + k] : 0.f;
  bp[idx] = (_Float16)v;
}

// ---------------------------------------------------------------------------
// 3) scores[b,t] = sum_h tanh(qb[b,h] + (enc[b,t,:]@Ua.T)[h]) * Va[h]
//    B-pack is staged in LDS once per block; inner loop = ds_load_b128 +
//    7 back-to-back v_wmma_f32_16x16x32_f16, VMEM pipe reserved for the
//    HBM-bound A-fragment stream of encoder_output.
// ---------------------------------------------------------------------------
__global__ __launch_bounds__(256) void k_scores(
    const float* __restrict__ enc, const _Float16* __restrict__ bpack,
    const float* __restrict__ qb, const float* __restrict__ Va,
    float* __restrict__ scores) {
  __shared__ uint4 ldsb[BPACK_VEC4];      // 93184 B of B fragments
  int b     = blockIdx.x >> 2;
  int chunk = blockIdx.x & 3;
  int tid   = threadIdx.x;
  int wave  = tid >> 5;
  int lane  = tid & 31;
  int lrow  = lane & 15;
  int lhalf = lane >> 4;

  // cooperative stage: global (L2-resident) -> LDS
  const uint4* bg = (const uint4*)bpack;
  for (int i = tid; i < BPACK_VEC4; i += 256) ldsb[i] = bg[i];
  __syncthreads();
  const v16h* blds = (const v16h*)ldsb;

  for (int tt = wave; tt < 32; tt += 8) {   // 32 tiles of 16 per chunk
    int t0 = chunk * 512 + tt * 16;
    const float* encrow = enc + ((size_t)b * T_ + t0 + lrow) * H_;

    if (tt + 8 < 32) {  // prefetch this lane's row of the next tile
      const float* pref = enc + ((size_t)b * T_ + t0 + 128 + lrow) * H_;
      __builtin_prefetch(pref, 0, 1);
    }

    // A fragments (16x32 f16): lanes 0-15 hold K 0-15 of row lrow,
    // lanes 16-31 hold K 16-31 of the same row.
    v16h a[NK_];
#pragma unroll
    for (int kk = 0; kk < NK_; ++kk) {
      int k0 = 32 * kk + (lhalf ? 16 : 0);
      int k1 = k0 + 8;
#pragma unroll
      for (int j = 0; j < 8; ++j) {
        int ka = k0 + j, kb = k1 + j;
        a[kk][j]     = (ka < H_) ? (_Float16)encrow[ka] : (_Float16)0.f;
        a[kk][j + 8] = (kb < H_) ? (_Float16)encrow[kb] : (_Float16)0.f;
      }
    }

    float sacc[8];
#pragma unroll
    for (int m = 0; m < 8; ++m) sacc[m] = 0.f;

#pragma unroll 1
    for (int nt = 0; nt < NT_; ++nt) {
      const v16h* bl = blds + (nt * NK_) * 32 + lane;
      v16h bf[NK_];
#pragma unroll
      for (int kk = 0; kk < NK_; ++kk) bf[kk] = bl[kk * 32];  // ds_load
      v8f c = {};
#pragma unroll
      for (int kk = 0; kk < NK_; ++kk) {
        c = __builtin_amdgcn_wmma_f32_16x16x32_f16(
            false, a[kk], false, bf[kk], (short)0, c, false, false);
      }
      // C/D layout: lanes 0-15 rows 0-7, lanes 16-31 rows 8-15; N = lrow+16*nt
      int h = nt * 16 + lrow;
      if (h < H_) {
        float qv = qb[b * H_ + h];
        float vv = Va[h];
#pragma unroll
        for (int m = 0; m < 8; ++m) sacc[m] += fast_tanh(c[m] + qv) * vv;
      }
    }
    // reduce over the 16 lanes of each half (sum over h columns)
#pragma unroll
    for (int m = 0; m < 8; ++m) {
      sacc[m] += __shfl_xor(sacc[m], 1, 16);
      sacc[m] += __shfl_xor(sacc[m], 2, 16);
      sacc[m] += __shfl_xor(sacc[m], 4, 16);
      sacc[m] += __shfl_xor(sacc[m], 8, 16);
    }
    if (lrow == 0) {
      float* sp = scores + (size_t)b * T_ + t0 + (lhalf ? 8 : 0);
#pragma unroll
      for (int m = 0; m < 8; ++m) sp[m] = sacc[m];
    }
  }
}

// ---------------------------------------------------------------------------
// 4) softmax over T, then context[b,h] = sum_t attn[b,t]*enc[b,t,h]
// ---------------------------------------------------------------------------
__global__ __launch_bounds__(256) void k_softmax_ctx(
    const float* __restrict__ enc, const float* __restrict__ scoresg,
    float* __restrict__ ctx) {
  __shared__ float s[T_];
  __shared__ float red[256];
  int b = blockIdx.x, tid = threadIdx.x;
  const float* sp = scoresg + (size_t)b * T_;
  float mx = -3.4e38f;
  for (int t = tid; t < T_; t += 256) { float v = sp[t]; s[t] = v; mx = fmaxf(mx, v); }
  red[tid] = mx; __syncthreads();
  for (int st = 128; st > 0; st >>= 1) {
    if (tid < st) red[tid] = fmaxf(red[tid], red[tid + st]);
    __syncthreads();
  }
  mx = red[0];
  __syncthreads();
  float ps = 0.f;
  for (int t = tid; t < T_; t += 256) { float e = __expf(s[t] - mx); s[t] = e; ps += e; }
  red[tid] = ps; __syncthreads();
  for (int st = 128; st > 0; st >>= 1) {
    if (tid < st) red[tid] += red[tid + st];
    __syncthreads();
  }
  float inv = 1.f / red[0];
  __syncthreads();
  if (tid < H_) {
    const float* ep = enc + (size_t)b * T_ * H_ + tid;
    float acc = 0.f;
#pragma unroll 8
    for (int t = 0; t < T_; ++t) acc += s[t] * ep[(size_t)t * H_];
    ctx[b * H_ + tid] = acc * inv;
  }
}

// ---------------------------------------------------------------------------
// 5) 5 LSTM+MLP decoder steps (per-batch block, sequential over steps).
//    Only the scalar feedback y changes between steps; the context GEMV,
//    c_prev (=c0) and h_pre are hoisted.
// ---------------------------------------------------------------------------
__global__ __launch_bounds__(256) void k_decoder(
    const float* __restrict__ x, const float* __restrict__ c0,
    const float* __restrict__ ctxg, const float* __restrict__ hpre,
    const float* __restrict__ Wih, const float* __restrict__ bih,
    const float* __restrict__ W1, const float* __restrict__ b1,
    const float* __restrict__ W2, const float* __restrict__ b2,
    const float* __restrict__ W3, const float* __restrict__ b3,
    float* __restrict__ out) {
  __shared__ float ctx[H_];
  __shared__ float gconst[H4_];
  __shared__ float wih0[H4_];
  __shared__ float hrelu[H_];
  __shared__ float o1[100];
  __shared__ float o2[64];
  __shared__ float ybc;
  int b = blockIdx.x, tid = threadIdx.x;
  for (int k = tid; k < H_; k += 256) ctx[k] = ctxg[b * H_ + k];
  __syncthreads();
  for (int j = tid; j < H4_; j += 256) {
    const float* w = Wih + (size_t)j * (H_ + 1);
    wih0[j] = w[0];
    float acc = bih[j] + hpre[b * H4_ + j];
    for (int k = 0; k < H_; ++k) acc += ctx[k] * w[1 + k];
    gconst[j] = acc;
  }
  __syncthreads();
  float y = x[b];
  for (int s = 0; s < 5; ++s) {
    if (tid < H_) {
      int h = tid;
      float gi = gconst[h]       + y * wih0[h];
      float gf = gconst[200 + h] + y * wih0[200 + h];
      float gg = gconst[400 + h] + y * wih0[400 + h];
      float go = gconst[600 + h] + y * wih0[600 + h];
      float cn = fast_sigmoid(gf) * c0[b * H_ + h] + fast_sigmoid(gi) * fast_tanh(gg);
      float hn = fast_sigmoid(go) * fast_tanh(cn);
      hrelu[h] = fmaxf(hn, 0.f);
    }
    __syncthreads();
    if (tid < 100) {
      float acc = b1[tid];
      const float* w = W1 + tid * H_;
      for (int k = 0; k < H_; ++k) acc += hrelu[k] * w[k];
      o1[tid] = fmaxf(acc, 0.f);
    }
    __syncthreads();
    if (tid < 50) {
      float acc = b2[tid];
      const float* w = W2 + tid * 100;
      for (int k = 0; k < 100; ++k) acc += o1[k] * w[k];
      o2[tid] = fmaxf(acc, 0.f);
    }
    __syncthreads();
    if (tid == 0) {
      float acc = b3[0];
      for (int k = 0; k < 50; ++k) acc += o2[k] * W3[k];
      out[b * 5 + s] = acc;
      ybc = acc;
    }
    __syncthreads();
    y = ybc;
  }
}

// ---------------------------------------------------------------------------
extern "C" void kernel_launch(void* const* d_in, const int* in_sizes, int n_in,
                              void* d_out, int out_size, void* d_ws, size_t ws_size,
                              hipStream_t stream) {
  const float* x   = (const float*)d_in[0];
  const float* h0  = (const float*)d_in[1];
  const float* c0  = (const float*)d_in[2];
  const float* enc = (const float*)d_in[3];
  const float* Wa  = (const float*)d_in[4];
  const float* ba  = (const float*)d_in[5];
  const float* Ua  = (const float*)d_in[6];
  const float* bua = (const float*)d_in[7];
  const float* Va  = (const float*)d_in[8];
  // d_in[9] = bva: softmax is shift-invariant, folds away (and is zero)
  const float* Wih = (const float*)d_in[10];
  const float* Whh = (const float*)d_in[11];
  const float* bih = (const float*)d_in[12];
  const float* bhh = (const float*)d_in[13];
  const float* W1  = (const float*)d_in[14];
  const float* b1  = (const float*)d_in[15];
  const float* W2  = (const float*)d_in[16];
  const float* b2  = (const float*)d_in[17];
  const float* W3  = (const float*)d_in[18];
  const float* b3  = (const float*)d_in[19];

  char* ws = (char*)d_ws;
  _Float16* bpack = (_Float16*)(ws + WS_BPACK);
  float* qb     = (float*)(ws + WS_QB);
  float* hpre   = (float*)(ws + WS_HPRE);
  float* scores = (float*)(ws + WS_SCORES);
  float* ctx    = (float*)(ws + WS_CTX);
  float* out    = (float*)d_out;

  k_prep<<<B_, 256, 0, stream>>>(h0, Wa, ba, bua, Whh, bhh, qb, hpre);
  k_pack<<<(NT_ * NK_ * 32 * 16 + 255) / 256, 256, 0, stream>>>(Ua, bpack);
  k_scores<<<B_ * 4, 256, 0, stream>>>(enc, bpack, qb, Va, scores);
  k_softmax_ctx<<<B_, 256, 0, stream>>>(enc, scores, ctx);
  k_decoder<<<B_, 256, 0, stream>>>(x, c0, ctx, hpre, Wih, bih,
                                    W1, b1, W2, b2, W3, b3, out);
}
